// MultiHeadAttention_55276229099598
// MI455X (gfx1250) — compile-verified, matrix operand order checked by reading, not run on hardware
//
#include <hip/hip_runtime.h>
#include <hip/hip_bf16.h>
#include <cstdint>
#include <cstddef>

// ---------------- problem constants ----------------
#define B_     2
#define S_     2048
#define D_     1024
#define H_     16
#define DEPTH_ 64
#define ROWS_  (B_ * S_)   // 4096

typedef __attribute__((ext_vector_type(16))) _Float16 v16h;
typedef __attribute__((ext_vector_type(8)))  float    v8f;
typedef __attribute__((ext_vector_type(4)))  unsigned int v4ui;
typedef __attribute__((ext_vector_type(8)))  int      v8i;
typedef __attribute__((ext_vector_type(4)))  int      v4i;

#define LOG2E_OVER_8 0.18033688011112042f   // log2(e)/sqrt(DEPTH)

__device__ __forceinline__ v8f wmma_f16(v16h a, v16h b, v8f c) {
  return __builtin_amdgcn_wmma_f32_16x16x32_f16(false, a, false, b, (short)0, c,
                                                false, false);
}

// ---- Tensor Data Mover: 2-D tile (f16 elements) global -> LDS ----
// D# packing per CDNA5 ISA (groups 2/3 zero => 2-D).
__device__ __forceinline__ void tdm_load_2d_f16(uint32_t lds_off,
                                                const void* gptr,
                                                uint32_t tile_d0,   // elems/row
                                                uint32_t tile_d1,   // rows
                                                uint32_t tensor_d0,
                                                uint32_t tensor_d1,
                                                uint32_t stride_el) {
  uint64_t ga = (uint64_t)(uintptr_t)gptr;
  v4ui g0;
  g0.x = 1u;                                           // count=1 (user mode)
  g0.y = lds_off;                                      // lds_addr (bytes)
  g0.z = (uint32_t)ga;                                 // global_addr[31:0]
  g0.w = (uint32_t)((ga >> 32) & 0x1FFFFFFu) | (2u << 30);  // addr[56:32]|type=2
  v8i g1;
  g1[0] = (int)(1u << 16);                             // data_size=1 (2 bytes)
  g1[1] = (int)(tensor_d0 << 16);                      // tensor_dim0[15:0]
  g1[2] = (int)((tensor_d0 >> 16) | (tensor_d1 << 16));
  g1[3] = (int)((tensor_d1 >> 16) | (tile_d0 << 16));  // tile_dim0
  g1[4] = (int)tile_d1;                                // tile_dim1 (dim2=0)
  g1[5] = (int)stride_el;                              // dim0_stride[31:0]
  g1[6] = 0;
  g1[7] = 0;
  v4i z4 = {0, 0, 0, 0};
#if defined(__clang_major__) && (__clang_major__ >= 23)
  v8i z8 = {0, 0, 0, 0, 0, 0, 0, 0};
  __builtin_amdgcn_tensor_load_to_lds(g0, g1, z4, z4, z8, 0);
#else
  __builtin_amdgcn_tensor_load_to_lds(g0, g1, z4, z4, 0);
#endif
}

__device__ __forceinline__ uint32_t lds_off_of(const void* p) {
  // generic LDS pointer: low 32 bits are the LDS byte offset (ISA §10.2)
  return (uint32_t)(uintptr_t)p;
}

// ---------------- fp32 -> f16 cast ----------------
__global__ void cast_f32_to_f16(const float* __restrict__ in,
                                _Float16* __restrict__ out, int n4) {
  int i = (blockIdx.x * blockDim.x + threadIdx.x) * 4;
  if (i < n4 * 4) {
    float4 v = *(const float4*)(in + i);
    out[i + 0] = (_Float16)v.x; out[i + 1] = (_Float16)v.y;
    out[i + 2] = (_Float16)v.z; out[i + 3] = (_Float16)v.w;
  }
}

// ---------------- W (K,N) f32 -> Wt (N,K) f16 ----------------
__global__ void transpose_cast(const float* __restrict__ W,
                               _Float16* __restrict__ Wt) {
  __shared__ float tile[32][33];
  int t = threadIdx.x;
  int x = t & 31, y = t >> 5;
  int bx = blockIdx.x * 32, by = blockIdx.y * 32;
  #pragma unroll
  for (int k = 0; k < 4; ++k)
    tile[y + 8 * k][x] = W[(size_t)(by + y + 8 * k) * D_ + bx + x];
  __syncthreads();
  #pragma unroll
  for (int k = 0; k < 4; ++k)
    Wt[(size_t)(bx + y + 8 * k) * D_ + by + x] = (_Float16)tile[x][y + 8 * k];
}

// ---------------- WMMA GEMM with TDM-staged operands ----------------
// C = A(MxK f16) * Bt(NxK f16)^T + bias.  Block 256 (8 waves), WG tile
// 128x64, per-wave 2x2 16x16 tiles.  64-wide K slices are DMA'd into LDS by
// the Tensor Data Mover (double buffered, wave 0 issues, TENSORcnt-paced):
// 8 WMMAs per barrier pair.
template <bool WF32, bool WF16>
__global__ void gemm_bias(const _Float16* __restrict__ A,
                          const _Float16* __restrict__ Bt,
                          const float* __restrict__ bias,
                          float* __restrict__ Cf, _Float16* __restrict__ Ch,
                          int M, int N, int K) {
  __shared__ _Float16 ldsA[2][128 * 64];   // 32 KB
  __shared__ _Float16 ldsB[2][64 * 64];    // 16 KB
  const int tid = threadIdx.x;
  const int w = tid >> 5, lane = tid & 31;
  const int l16 = lane & 15, lh = lane >> 4;
  const int m0 = blockIdx.y * 128, n0 = blockIdx.x * 64;
  const int mg = w & 3, ng = w >> 2;
  const int nIter = K >> 6;                // 64-wide K slices

  if (w == 0) {
    tdm_load_2d_f16(lds_off_of(&ldsA[0][0]), A + (size_t)m0 * K, 64, 128,
                    (uint32_t)K, (uint32_t)M, (uint32_t)K);
    tdm_load_2d_f16(lds_off_of(&ldsB[0][0]), Bt + (size_t)n0 * K, 64, 64,
                    (uint32_t)K, (uint32_t)N, (uint32_t)K);
  }

  v8f acc[2][2] = {};
  for (int it = 0; it < nIter; ++it) {
    const int cur = it & 1;
    if (w == 0) {
      if (it + 1 < nIter) {
        const int nxt = cur ^ 1;
        tdm_load_2d_f16(lds_off_of(&ldsA[nxt][0]),
                        A + (size_t)m0 * K + (it + 1) * 64, 64, 128,
                        (uint32_t)K, (uint32_t)M, (uint32_t)K);
        tdm_load_2d_f16(lds_off_of(&ldsB[nxt][0]),
                        Bt + (size_t)n0 * K + (it + 1) * 64, 64, 64,
                        (uint32_t)K, (uint32_t)N, (uint32_t)K);
        __builtin_amdgcn_s_wait_tensorcnt(2);  // current pair complete
      } else {
        __builtin_amdgcn_s_wait_tensorcnt(0);
      }
    }
    __syncthreads();
    #pragma unroll
    for (int ks = 0; ks < 64; ks += 32) {
      v16h a[2], bf[2];
      #pragma unroll
      for (int mi = 0; mi < 2; ++mi)
        a[mi] = *(const v16h*)(
            &ldsA[cur][((2 * mg + mi) * 16 + l16) * 64 + ks + lh * 16]);
      #pragma unroll
      for (int ni = 0; ni < 2; ++ni)
        bf[ni] = *(const v16h*)(
            &ldsB[cur][((2 * ng + ni) * 16 + l16) * 64 + ks + lh * 16]);
      #pragma unroll
      for (int mi = 0; mi < 2; ++mi)
        #pragma unroll
        for (int ni = 0; ni < 2; ++ni)
          acc[mi][ni] = wmma_f16(a[mi], bf[ni], acc[mi][ni]);
    }
    __syncthreads();  // buffer `cur` is overwritten next iteration
  }

  #pragma unroll
  for (int mi = 0; mi < 2; ++mi)
    #pragma unroll
    for (int ni = 0; ni < 2; ++ni)
      #pragma unroll
      for (int i = 0; i < 8; ++i) {
        int row = m0 + (2 * mg + mi) * 16 + i + lh * 8;
        int col = n0 + (2 * ng + ni) * 16 + l16;
        float v = acc[mi][ni][i] + bias[col];
        if (WF32) Cf[(size_t)row * N + col] = v;
        if (WF16) Ch[(size_t)row * N + col] = (_Float16)v;
      }
}

// ---------------- LayerNorm (+ per-head epilogues) ----------------
template <int MODE>  // 0=q: softmax+ent, 1=k: softmax->log, 2=v: transpose
__global__ void ln_kernel(const float* __restrict__ X, const float* __restrict__ g,
                          const float* __restrict__ bb, _Float16* __restrict__ out,
                          float* __restrict__ ent_out) {
  const int row = blockIdx.x;
  const int t = threadIdx.x;
  const int w = t >> 5, lane = t & 31;
  __shared__ float y[D_];
  __shared__ float r1[8], r2[8];
  __shared__ float muS, rvS;
  float lv[4], s1 = 0.f, s2 = 0.f;
  #pragma unroll
  for (int j = 0; j < 4; ++j) {
    float x = X[(size_t)row * D_ + t + 256 * j];
    lv[j] = x; s1 += x; s2 += x * x;
  }
  #pragma unroll
  for (int off = 16; off; off >>= 1) {
    s1 += __shfl_xor(s1, off, 32);
    s2 += __shfl_xor(s2, off, 32);
  }
  if (lane == 0) { r1[w] = s1; r2[w] = s2; }
  __syncthreads();
  if (t == 0) {
    float a = 0.f, c = 0.f;
    for (int i = 0; i < 8; ++i) { a += r1[i]; c += r2[i]; }
    float mu = a * (1.0f / D_);
    muS = mu;
    rvS = rsqrtf(c * (1.0f / D_) - mu * mu + 1e-6f);
  }
  __syncthreads();
  #pragma unroll
  for (int j = 0; j < 4; ++j) {
    int d = t + 256 * j;
    y[d] = (lv[j] - muS) * rvS * g[d] + bb[d];
  }
  __syncthreads();

  if (MODE == 2) {
    int bidx = row >> 11, s = row & (S_ - 1);
    #pragma unroll
    for (int j = 0; j < 4; ++j) {
      int d = t + 256 * j;
      int h = d >> 6, dd = d & 63;
      out[((size_t)((bidx * H_ + h) * DEPTH_ + dd)) * S_ + s] = (_Float16)y[d];
    }
  } else {
    int grp = t >> 4, r = t & 15, base = grp * 64;
    float xv[4], m = -1e30f;
    #pragma unroll
    for (int j = 0; j < 4; ++j) { xv[j] = y[base + r + 16 * j]; m = fmaxf(m, xv[j]); }
    #pragma unroll
    for (int off = 8; off; off >>= 1) m = fmaxf(m, __shfl_xor(m, off, 16));
    float p[4], s = 0.f;
    #pragma unroll
    for (int j = 0; j < 4; ++j) { p[j] = __expf(xv[j] - m); s += p[j]; }
    #pragma unroll
    for (int off = 8; off; off >>= 1) s += __shfl_xor(s, off, 16);
    float inv = __builtin_amdgcn_rcpf(s);
    if (MODE == 0) {
      float e = 0.f;
      #pragma unroll
      for (int j = 0; j < 4; ++j) {
        float prob = p[j] * inv;
        out[(size_t)row * D_ + base + r + 16 * j] = (_Float16)prob;
        e += prob * __logf(prob);
      }
      #pragma unroll
      for (int off = 8; off; off >>= 1) e += __shfl_xor(e, off, 16);
      if (r == 0) {
        int bidx = row >> 11, ss = row & (S_ - 1);
        ent_out[(size_t)(bidx * H_ + grp) * S_ + ss] = e;
      }
    } else {
      #pragma unroll
      for (int j = 0; j < 4; ++j)
        out[(size_t)row * D_ + base + r + 16 * j] =
            (_Float16)__logf(p[j] * inv + 2e-8f);
    }
  }
}

// ---------------- fused scores -> csch softmax -> attn@v ----------------
// grid = (S/32, B*H); block = 256 (8 waves).  32x2048 fp32 numerator strip in
// dynamic LDS (256 KB); log(k_prob) key tiles and (after the key loop) vt
// tiles share the same 2x16 KB TDM-staged static LDS buffers.
__global__ void attn_kernel(const _Float16* __restrict__ qprob,
                            const _Float16* __restrict__ logk,
                            const _Float16* __restrict__ vt,
                            const float* __restrict__ ent,
                            float* __restrict__ attn_out,
                            _Float16* __restrict__ out_pre) {
  __shared__ _Float16 ldsK[2][128 * DEPTH_];   // 2 x 16 KB (keys, then vt)
  extern __shared__ char smem[];
  float* numer = (float*)smem;                 // 32 * 2048 fp32 = 256 KB
  float* entS = numer + 32 * S_;               // 32
  float* rowsum = entS + 32;                   // 32
  const int tid = threadIdx.x;
  const int w = tid >> 5, lane = tid & 31;
  const int l16 = lane & 15, lh = lane >> 4;
  const int q0 = blockIdx.x * 32;
  const int bh = blockIdx.y;
  const int b = bh >> 4, h = bh & 15;

  if (tid < 32) {
    entS[tid] = ent[(size_t)bh * S_ + q0 + tid];
    rowsum[tid] = 0.f;
  }
  if (w == 0)   // stage first 128-key tile of log(k_prob) for this head
    tdm_load_2d_f16(lds_off_of(&ldsK[0][0]),
                    logk + (size_t)(b * S_) * D_ + h * DEPTH_,
                    DEPTH_, 128, DEPTH_, S_, D_);
  __syncthreads();

  const int mt = w & 1;
  const int ntg = w >> 1;
  // q_prob A fragments: registers for the entire key loop
  const int qrow = q0 + mt * 16 + l16;
  const _Float16* qb = qprob + (size_t)(b * S_ + qrow) * D_ + h * DEPTH_ + lh * 16;
  const v16h a0 = *(const v16h*)(qb);
  const v16h a1 = *(const v16h*)(qb + 32);
  float er[8];
  #pragma unroll
  for (int i = 0; i < 8; ++i) er[i] = entS[mt * 16 + i + lh * 8];

  float rs[8] = {};
  for (int n0 = 0; n0 < S_; n0 += 128) {
    const int cur = (n0 >> 7) & 1;
    if (w == 0) {
      if (n0 + 128 < S_) {
        tdm_load_2d_f16(lds_off_of(&ldsK[cur ^ 1][0]),
                        logk + (size_t)(b * S_ + n0 + 128) * D_ + h * DEPTH_,
                        DEPTH_, 128, DEPTH_, S_, D_);
        __builtin_amdgcn_s_wait_tensorcnt(1);  // tile `cur` landed
      } else {
        __builtin_amdgcn_s_wait_tensorcnt(0);
      }
    }
    __syncthreads();
    #pragma unroll
    for (int nt2 = 0; nt2 < 2; ++nt2) {
      const int nl = (ntg * 2 + nt2) * 16;
      const _Float16* kb = &ldsK[cur][(nl + l16) * DEPTH_ + lh * 16];
      const v16h b0 = *(const v16h*)(kb);
      const v16h b1 = *(const v16h*)(kb + 32);
      v8f acc = {};
      acc = wmma_f16(a0, b0, acc);
      acc = wmma_f16(a1, b1, acc);
      #pragma unroll
      for (int i = 0; i < 8; ++i) {
        const int rl = mt * 16 + i + lh * 8;
        const int col = n0 + nl + l16;
        float e = __builtin_amdgcn_exp2f((er[i] - acc[i]) * LOG2E_OVER_8);
        float csch = 2.0f * e * __builtin_amdgcn_rcpf(e * e - 1.0f);
        numer[rl * S_ + col] = csch;
        rs[i] += csch;
      }
    }
    __syncthreads();  // done reading ldsK[cur] before overwrite
  }

  // prefetch first vt tile into ldsK[0]; DMA overlaps the softmax reduction
  if (w == 0)
    tdm_load_2d_f16(lds_off_of(&ldsK[0][0]), vt + (size_t)(bh * DEPTH_) * S_,
                    128, DEPTH_, S_, DEPTH_, S_);

  #pragma unroll
  for (int i = 0; i < 8; ++i) {
    #pragma unroll
    for (int off = 8; off; off >>= 1) rs[i] += __shfl_xor(rs[i], off, 16);
    if (l16 == 0) atomicAdd(&rowsum[mt * 16 + i + lh * 8], rs[i]);
  }
  __syncthreads();
  if (tid < 32) rowsum[tid] = __builtin_amdgcn_rcpf(rowsum[tid]);
  __syncthreads();

  // normalize in LDS and stream fp32 attn to d_out (single HBM write)
  float* attn_base = attn_out + ((size_t)bh * S_ + q0) * S_;
  for (int idx = tid; idx < 32 * S_; idx += 256) {
    int rl = idx >> 11;
    float v = numer[idx] * rowsum[rl];
    numer[idx] = v;
    attn_base[idx] = v;
  }
  __syncthreads();

  // out(32x64) = attn(32x2048, LDS) @ v(2048x64, TDM-staged 128-key tiles)
  const int nt = w >> 1;
  v8f acc = {};
  const float* ab = numer + (mt * 16 + l16) * S_ + lh * 16;
  for (int it = 0; it < S_ / 128; ++it) {
    const int cur = it & 1;
    if (w == 0) {
      if (it + 1 < S_ / 128) {
        tdm_load_2d_f16(lds_off_of(&ldsK[cur ^ 1][0]),
                        vt + (size_t)(bh * DEPTH_) * S_ + (it + 1) * 128,
                        128, DEPTH_, S_, DEPTH_, S_);
        __builtin_amdgcn_s_wait_tensorcnt(1);
      } else {
        __builtin_amdgcn_s_wait_tensorcnt(0);
      }
    }
    __syncthreads();
    #pragma unroll
    for (int ks = 0; ks < 128; ks += 32) {
      const int k0 = it * 128 + ks;
      float4 f0 = *(const float4*)(ab + k0 + 0);
      float4 f1 = *(const float4*)(ab + k0 + 4);
      float4 f2 = *(const float4*)(ab + k0 + 8);
      float4 f3 = *(const float4*)(ab + k0 + 12);
      v16h a;
      a[0] = (_Float16)f0.x;  a[1] = (_Float16)f0.y;
      a[2] = (_Float16)f0.z;  a[3] = (_Float16)f0.w;
      a[4] = (_Float16)f1.x;  a[5] = (_Float16)f1.y;
      a[6] = (_Float16)f1.z;  a[7] = (_Float16)f1.w;
      a[8] = (_Float16)f2.x;  a[9] = (_Float16)f2.y;
      a[10] = (_Float16)f2.z; a[11] = (_Float16)f2.w;
      a[12] = (_Float16)f3.x; a[13] = (_Float16)f3.y;
      a[14] = (_Float16)f3.z; a[15] = (_Float16)f3.w;
      v16h bf = *(const v16h*)(&ldsK[cur][(nt * 16 + l16) * 128 + ks + lh * 16]);
      acc = wmma_f16(a, bf, acc);
    }
    __syncthreads();
  }
  #pragma unroll
  for (int i = 0; i < 8; ++i) {
    int q = q0 + mt * 16 + i + lh * 8;
    int d = nt * 16 + l16;
    out_pre[(size_t)(b * S_ + q) * D_ + h * DEPTH_ + d] = (_Float16)acc[i];
  }
}

// ---------------- host side ----------------
extern "C" void kernel_launch(void* const* d_in, const int* in_sizes, int n_in,
                              void* d_out, int out_size, void* d_ws,
                              size_t ws_size, hipStream_t stream) {
  const float* q_in = (const float*)d_in[0];
  const float* k_in = (const float*)d_in[1];
  const float* Wq = (const float*)d_in[2];
  const float* bq = (const float*)d_in[3];
  const float* Wk = (const float*)d_in[4];
  const float* bk = (const float*)d_in[5];
  const float* Wv = (const float*)d_in[6];
  const float* bv = (const float*)d_in[7];
  const float* Wo = (const float*)d_in[8];
  const float* bo = (const float*)d_in[9];
  const float* g1 = (const float*)d_in[10];
  const float* b1 = (const float*)d_in[11];
  const float* g2 = (const float*)d_in[12];
  const float* b2 = (const float*)d_in[13];
  const float* g3 = (const float*)d_in[14];
  const float* b3 = (const float*)d_in[15];

  float* out_f = (float*)d_out;                        // (B,S,D) fp32
  float* attn_f = out_f + (size_t)ROWS_ * D_;          // (B,H,S,S) fp32

  char* ws = (char*)d_ws;
  auto alloc = [&](size_t bytes) -> char* {
    char* p = ws;
    ws += (bytes + 255) & ~(size_t)255;
    return p;
  };
  const size_t ND = (size_t)ROWS_ * D_;
  const size_t WW = (size_t)D_ * D_;
  _Float16* qinH = (_Float16*)alloc(ND * 2);
  _Float16* kinH = (_Float16*)alloc(ND * 2);
  _Float16* WqT = (_Float16*)alloc(WW * 2);
  _Float16* WkT = (_Float16*)alloc(WW * 2);
  _Float16* WvT = (_Float16*)alloc(WW * 2);
  _Float16* WoT = (_Float16*)alloc(WW * 2);
  _Float16* t1H = (_Float16*)alloc(ND * 2);
  _Float16* k1H = (_Float16*)alloc(ND * 2);
  float* qpreF = (float*)alloc(ND * 4);
  float* kpreF = (float*)alloc(ND * 4);
  float* vpreF = (float*)alloc(ND * 4);
  _Float16* qprobH = (_Float16*)alloc(ND * 2);
  _Float16* logkH = (_Float16*)alloc(ND * 2);
  _Float16* vtH = (_Float16*)alloc(ND * 2);
  float* entF = (float*)alloc((size_t)B_ * H_ * S_ * 4);
  _Float16* outpreH = (_Float16*)alloc(ND * 2);

  const dim3 blk(256);
  const dim3 castGrid((unsigned)(ND / (256 * 4)));
  const dim3 trGrid(32, 32);
  const dim3 gemmGrid(D_ / 64, ROWS_ / 128);           // (16, 32)
  const dim3 lnGrid(ROWS_);
  const dim3 attnGrid(S_ / 32, B_ * H_);               // (64, 32)
  const int ATTN_SMEM = 32 * S_ * 4 + 64 * 4;          // 262,400 bytes dynamic

  (void)hipFuncSetAttribute((const void*)attn_kernel,
                            hipFuncAttributeMaxDynamicSharedMemorySize,
                            ATTN_SMEM);

  cast_f32_to_f16<<<castGrid, blk, 0, stream>>>(q_in, qinH, (int)(ND / 4));
  cast_f32_to_f16<<<castGrid, blk, 0, stream>>>(k_in, kinH, (int)(ND / 4));
  transpose_cast<<<trGrid, blk, 0, stream>>>(Wq, WqT);
  transpose_cast<<<trGrid, blk, 0, stream>>>(Wk, WkT);
  transpose_cast<<<trGrid, blk, 0, stream>>>(Wv, WvT);
  transpose_cast<<<trGrid, blk, 0, stream>>>(Wo, WoT);

  gemm_bias<false, true><<<gemmGrid, blk, 0, stream>>>(
      qinH, WqT, bq, nullptr, t1H, ROWS_, D_, D_);
  gemm_bias<true, false><<<gemmGrid, blk, 0, stream>>>(
      t1H, WqT, bq, qpreF, nullptr, ROWS_, D_, D_);
  gemm_bias<false, true><<<gemmGrid, blk, 0, stream>>>(
      kinH, WkT, bk, nullptr, k1H, ROWS_, D_, D_);
  gemm_bias<true, false><<<gemmGrid, blk, 0, stream>>>(
      k1H, WkT, bk, kpreF, nullptr, ROWS_, D_, D_);
  gemm_bias<true, false><<<gemmGrid, blk, 0, stream>>>(
      k1H, WvT, bv, vpreF, nullptr, ROWS_, D_, D_);

  ln_kernel<0><<<lnGrid, blk, 0, stream>>>(qpreF, g1, b1, qprobH, entF);
  ln_kernel<1><<<lnGrid, blk, 0, stream>>>(kpreF, g2, b2, logkH, nullptr);
  ln_kernel<2><<<lnGrid, blk, 0, stream>>>(vpreF, g3, b3, vtH, nullptr);

  attn_kernel<<<attnGrid, blk, ATTN_SMEM, stream>>>(qprobH, logkH, vtH, entF,
                                                    attn_f, outpreH);

  gemm_bias<true, false><<<gemmGrid, blk, 0, stream>>>(
      outpreH, WoT, bo, out_f, nullptr, ROWS_, D_, D_);
}
